// VandermondeConv_695784702184
// MI455X (gfx1250) — matrix operand order repaired and checked by reading, not compiled, not run on hardware
//
#include <hip/hip_runtime.h>
#include <math.h>

// ---------------------------------------------------------------------------
// Problem constants (from reference): B=16, H=256, L=4096, P=64
// y[b,h,:] = tanh( circconv(u[b,h,:], Re(k[h,:])) + D[h,h]*u[b,h,:] )
// k[h,l] = sum_p (C[h,p]*B_bar[p,h]) * Lambda_p^l
// Circulant matmul per h:  Y (L x 16) = K_h (L x L) * U (L x 16),
// K_h[l,t] = k_r[h, (l-t) mod L].
// We materialize the kernel TIME-REVERSED: kext_rev[h,i] = k_r[h,(2L-1-i) mod L]
// so that within a tile A[l',t'] = ks[t' - l' + const] ASCENDS in the WMMA K
// dimension -> contiguous LDS b128 loads, no per-lane reversal movs.
// ---------------------------------------------------------------------------

#define B_DIM 16
#define H_DIM 256
#define L_DIM 4096
#define P_DIM 64

#define KC     256          // K (t) chunk staged in LDS per iteration
#define MT     128          // l rows per workgroup (8 waves x 16)
#define NCHUNK (L_DIM / KC) // 16

typedef __attribute__((ext_vector_type(16))) __bf16 v16bf;
typedef __attribute__((ext_vector_type(8)))  float  v8f;

__device__ __forceinline__ unsigned short f32_to_bf16_u16(float f) {
  union { float f; unsigned u; } v; v.f = f;
  unsigned u = v.u;
  u += 0x7FFFu + ((u >> 16) & 1u);   // round-to-nearest-even
  return (unsigned short)(u >> 16);
}

// ---------------------------------------------------------------------------
// Phase 1: kext_rev[h, i] = bf16( Re( sum_p w[h,p] * Lambda_p^((2L-1-i) mod L) ) )
// for i in [0, 2L).  w = C .* B^T (complex).
// Lambda^l = e^{l ln r} (cos l*theta, sin l*theta): stable, |Lambda|<1 never
// inverted.  grid (2L/256, H), block 256.
// ---------------------------------------------------------------------------
__global__ void vand_kernel_build(const float* __restrict__ C,
                                  const float* __restrict__ Bb,
                                  const float* __restrict__ Lam,
                                  unsigned short* __restrict__ kext) {
  __shared__ float wr[P_DIM], wi[P_DIM], lr[P_DIM], th[P_DIM];
  const int h   = blockIdx.y;
  const int tid = threadIdx.x;
  if (tid < P_DIM) {
    const int p = tid;
    float cr = C[(h * P_DIM + p) * 2 + 0], ci = C[(h * P_DIM + p) * 2 + 1];
    float br = Bb[(p * H_DIM + h) * 2 + 0], bi = Bb[(p * H_DIM + h) * 2 + 1];
    wr[p] = cr * br - ci * bi;
    wi[p] = cr * bi + ci * br;
    float re = Lam[p * 2 + 0], im = Lam[p * 2 + 1];
    lr[p] = 0.5f * __logf(re * re + im * im);   // ln r
    th[p] = atan2f(im, re);                     // theta
  }
  __syncthreads();

  const int i  = blockIdx.x * 256 + tid;
  const int l  = (2 * L_DIM - 1 - i) & (L_DIM - 1);   // time-reversed index
  const float fl = (float)l;
  float acc = 0.0f;
#pragma unroll 4
  for (int p = 0; p < P_DIM; ++p) {
    float e = __expf(fl * lr[p]);
    float s, c;
    __sincosf(fl * th[p], &s, &c);
    acc += e * (wr[p] * c - wi[p] * s);
  }
  kext[(size_t)h * (2 * L_DIM) + i] = f32_to_bf16_u16(acc);
}

// ---------------------------------------------------------------------------
// Phase 2: per-h circulant GEMM with bf16 WMMA.
// grid (L/128, H), block 256 (8 waves).  Each wave owns one 16-row M tile.
// LDS: u tile 16x256 bf16 (f32->bf16 via VALU) + kernel slice 384 bf16
// (staged with global_load_async_to_lds_b32, tracked by ASYNCcnt; the source
// offset ibase = L-128-l0+t0 is 128-aligned by construction).
// A[l',t'] = ks[t' - l' + 127]  (ascending in K -> b128-friendly).
// Accumulate f32, epilogue adds diag(D)*u and tanh, float4 stores.
// ---------------------------------------------------------------------------
__global__ void __launch_bounds__(256)
vand_kernel_conv(const float* __restrict__ u,
                 const float* __restrict__ D,
                 const unsigned short* __restrict__ kext,
                 float* __restrict__ out) {
  __shared__ __align__(32) unsigned short u_lds[B_DIM * KC]; // 16 x 256 bf16
  __shared__ __align__(16) unsigned short ks[KC + MT];       // 384 bf16 slice

  const int h    = blockIdx.y;
  const int l0   = blockIdx.x * MT;
  const int tid  = threadIdx.x;
  const int wave = tid >> 5;
  const int lane = tid & 31;
  const int m    = lane & 15;   // A row within tile / C column (batch)
  const int hi   = lane >> 4;

  // staging roles: thread -> (batch ub, 16-wide segment useg)
  const int ub   = tid >> 4;
  const int useg = (tid & 15) * 16;
  const float* uRow = u + ((size_t)ub * H_DIM + h) * L_DIM;

  v8f acc = {0.f, 0.f, 0.f, 0.f, 0.f, 0.f, 0.f, 0.f};

  for (int chunk = 0; chunk < NCHUNK; ++chunk) {
    const int t0 = chunk * KC;
    __syncthreads();   // previous chunk's LDS reads complete

    // --- stage u[ub, h, t0..t0+255] -> bf16 LDS tile -----------------------
    {
      const float4* src = (const float4*)(uRow + t0 + useg);
      float tmp[16];
      *(float4*)(tmp + 0)  = src[0];
      *(float4*)(tmp + 4)  = src[1];
      *(float4*)(tmp + 8)  = src[2];
      *(float4*)(tmp + 12) = src[3];
      unsigned short* dst = u_lds + ub * KC + useg;
#pragma unroll
      for (int i = 0; i < 16; ++i) dst[i] = f32_to_bf16_u16(tmp[i]);
      if (chunk + 1 < NCHUNK)      // hint next chunk into cache
        __builtin_prefetch(uRow + t0 + KC + useg, 0, 0);
    }

    // --- stage kernel slice via async LDS copy (ASYNCcnt) ------------------
    // ks[d] = kext_rev[h, ibase + d], d in [0,384); ibase is 128-aligned.
    const int ibase = (L_DIM - MT) - l0 + t0;
    const unsigned short* kg = kext + (size_t)h * (2 * L_DIM) + ibase;
    if (tid < 192) {  // 192 dwords = 384 bf16
      unsigned long long ga = (unsigned long long)(uintptr_t)(kg + 2 * tid);
      unsigned lo = (unsigned)(uintptr_t)(&ks[2 * tid]);
      asm volatile("global_load_async_to_lds_b32 %0, %1, off"
                   :: "v"(lo), "v"(ga) : "memory");
    }
    asm volatile("s_wait_asynccnt 0" ::: "memory");
    __syncthreads();   // tiles visible to all waves

    // --- 8 x v_wmma_f32_16x16x32_bf16 over this K chunk --------------------
    // A index = (t' - l') + 127 = kc + K - (wave*16+m) + 127, ascending in K.
    const int abase = 127 - (wave * 16 + m);
#pragma unroll
    for (int kc = 0; kc < KC; kc += 32) {
      v16bf a;
#pragma unroll
      for (int e = 0; e < 16; ++e) {
        // A 16x32 bf16 lane layout: K = hi*16 + e (contiguous, matches B)
        const int K = hi * 16 + e;
        a[e] = __builtin_bit_cast(__bf16, ks[abase + kc + K]);
      }
      // B 32x16 bf16: lane n=m holds column N=m, K = hi*16 + e (contiguous)
      const v16bf b = *(const v16bf*)(u_lds + m * KC + kc + hi * 16);
      acc = __builtin_amdgcn_wmma_f32_16x16x32_bf16(
          false, a, false, b, (short)0, acc, false, false);
    }
  }

  // --- epilogue: + diag(D)*u, tanh, store ---------------------------------
  // C/D layout: lane n = m (batch), VGPR r -> M = r + 8*hi
  const float dd   = D[h * (H_DIM + 1)];
  const int  lbase = l0 + wave * 16 + hi * 8;
  const size_t off = ((size_t)m * H_DIM + h) * L_DIM + lbase;
  const float4 u0 = *(const float4*)(u + off);
  const float4 u1 = *(const float4*)(u + off + 4);
  float4 y0, y1;
  y0.x = tanhf(acc[0] + dd * u0.x);
  y0.y = tanhf(acc[1] + dd * u0.y);
  y0.z = tanhf(acc[2] + dd * u0.z);
  y0.w = tanhf(acc[3] + dd * u0.w);
  y1.x = tanhf(acc[4] + dd * u1.x);
  y1.y = tanhf(acc[5] + dd * u1.y);
  y1.z = tanhf(acc[6] + dd * u1.z);
  y1.w = tanhf(acc[7] + dd * u1.w);
  *(float4*)(out + off)     = y0;
  *(float4*)(out + off + 4) = y1;
}

// ---------------------------------------------------------------------------
extern "C" void kernel_launch(void* const* d_in, const int* in_sizes, int n_in,
                              void* d_out, int out_size, void* d_ws, size_t ws_size,
                              hipStream_t stream) {
  const float* u      = (const float*)d_in[0];  // (B,H,L)
  const float* C      = (const float*)d_in[1];  // (H,P,2)
  const float* Dm     = (const float*)d_in[2];  // (H,H)
  const float* B_bar  = (const float*)d_in[3];  // (P,H,2)
  const float* Lambda = (const float*)d_in[4];  // (P,2)
  // d_in[5] = powers (implicit arange, unused)
  float* out = (float*)d_out;

  unsigned short* kext = (unsigned short*)d_ws;  // H * 2L bf16 = 4 MB

  dim3 g1(2 * L_DIM / 256, H_DIM);
  vand_kernel_build<<<g1, 256, 0, stream>>>(C, B_bar, Lambda, kext);

  dim3 g2(L_DIM / MT, H_DIM);
  vand_kernel_conv<<<g2, 256, 0, stream>>>(u, Dm, kext, out);

  (void)in_sizes; (void)n_in; (void)out_size; (void)ws_size;
}
